// GATLayer_17660905521755
// MI455X (gfx1250) — compile-verified, matrix operand order checked by reading, not run on hardware
//
#include <hip/hip_runtime.h>
#include <hip/hip_bf16.h>

typedef __attribute__((ext_vector_type(16))) _Float16 v16h;
typedef __attribute__((ext_vector_type(8)))  float    v8f;

#define NN 8192
#define FF 128
#define ALPHA_SLOPE 0.2f

__device__ __forceinline__ float lrelu(float t) { return t > 0.f ? t : ALPHA_SLOPE * t; }

// ---------------------------------------------------------------------------
// Kernel 1: h = x*W (elementwise), f1 = h@a[:F], f2 = h@a[F:],
// and scatter h (f16) into the WMMA B-operand lane layout:
//   B tile (K=32 rows x N=16 cols): lane = laneHi*16 + n, element e holds
//   K = e + (e<8 ? 0 : 8) + 8*laneHi  (CDNA5 ISA 7.12.2, 16-bit operand map)
// Storage: hB[((kc*8 + c)*32 + lane)*16 + e], kc = j/32, c = n/16.
// ---------------------------------------------------------------------------
__global__ void gat_prep(const float* __restrict__ x, const float* __restrict__ W,
                         const float* __restrict__ a,
                         _Float16* __restrict__ hB,
                         float* __restrict__ f1, float* __restrict__ f2) {
    __shared__ float s1[FF], s2[FF];
    const int i = blockIdx.x;        // row (node) index = global K index j
    const int n = threadIdx.x;       // feature/column
    const float hv = x[(size_t)i * FF + n] * W[(size_t)i * FF + n];

    const int kc = i >> 5, kl = i & 31;
    const int hi = (kl >> 3) & 1;
    const int e  = (kl & 7) + (kl >= 16 ? 8 : 0);
    const int c  = n >> 4, laneLo = n & 15;
    const int lane = hi * 16 + laneLo;
    hB[(((size_t)(kc * 8 + c) * 32 + lane) << 4) + e] = (_Float16)hv;

    s1[n] = hv * a[n];
    s2[n] = hv * a[FF + n];
    __syncthreads();
    for (int off = FF / 2; off > 0; off >>= 1) {
        if (n < off) { s1[n] += s1[n + off]; s2[n] += s2[n + off]; }
        __syncthreads();
    }
    if (n == 0) { f1[i] = s1[0]; f2[i] = s2[0]; }
}

// ---------------------------------------------------------------------------
// Kernel 2: per-row online softmax stats (m_i, s_i) over masked
// e[i,j] = lrelu(f1[i]+f2[j]), fused with v[i] = sum_j adj[i,j]*W2[j].
// Single pass over the 256 MB adjacency matrix.
// ---------------------------------------------------------------------------
__global__ void gat_stats(const int* __restrict__ adj, const float* __restrict__ W2,
                          const float* __restrict__ f1, const float* __restrict__ f2,
                          float* __restrict__ mrow, float* __restrict__ srow,
                          float* __restrict__ vrow) {
    __shared__ float sm[256], ss[256], sv[256];
    const int i = blockIdx.x, t = threadIdx.x;
    const int* arow = adj + (size_t)i * NN;
    const float f1i = f1[i];
    float mloc = -INFINITY, sloc = 0.f, vloc = 0.f;
    for (int j = t; j < NN; j += 256) {
        const int av = arow[j];
        vloc += (float)av * W2[j];
        if (av > 0) {
            const float lt = lrelu(f1i + f2[j]);
            if (lt > mloc) { sloc = sloc * __expf(mloc - lt) + 1.f; mloc = lt; }
            else           { sloc += __expf(lt - mloc); }
        }
    }
    sm[t] = mloc; ss[t] = sloc; sv[t] = vloc;
    __syncthreads();
    for (int off = 128; off > 0; off >>= 1) {
        if (t < off) {
            const float m1 = sm[t], m2 = sm[t + off];
            const float mn = fmaxf(m1, m2);
            float sn = 0.f;
            if (mn != -INFINITY)
                sn = ss[t] * __expf(m1 - mn) + ss[t + off] * __expf(m2 - mn);
            sm[t] = mn; ss[t] = sn; sv[t] += sv[t + off];
        }
        __syncthreads();
    }
    if (t == 0) { mrow[i] = sm[0]; srow[i] = ss[0]; vrow[i] = sv[0]; }
}

// ---------------------------------------------------------------------------
// Kernel 3: h_prime = attention @ h via v_wmma_f32_16x16x32_f16.
// One wave per 16-row output stripe; 8 accumulators cover all 128 columns.
// Attention weights generated in-register per K-chunk of 32:
//   A[e] = adj>0 ? exp(lrelu(f1[i]+f2[j]) - m_i) / s_i : 0   (f16)
// A-operand K map (lane L, laneHi = L>=16): K = e + (e<8?0:8) + 8*laneHi,
// which is exactly two contiguous runs of 8 -> four int4/float4 gathers.
// ---------------------------------------------------------------------------
__global__ __launch_bounds__(32)
void gat_attn_gemm(const int* __restrict__ adj, const _Float16* __restrict__ hB,
                   const float* __restrict__ f1, const float* __restrict__ f2,
                   const float* __restrict__ mrow, const float* __restrict__ srow,
                   float* __restrict__ hprime) {
    const int lane   = threadIdx.x;
    const int laneLo = lane & 15;
    const int laneHi = lane >> 4;
    const int i0 = blockIdx.x * 16;
    const int i  = i0 + laneLo;             // A operand: lane holds row M = lane%16

    const float f1v  = f1[i];
    const float mv   = mrow[i];
    const float sv   = srow[i];
    const float invs = sv > 0.f ? 1.f / sv : 0.f;

    const int*  arow = adj + (size_t)i * NN;
    const v16h* bmat = (const v16h*)hB;

    const v8f zero = {0.f, 0.f, 0.f, 0.f, 0.f, 0.f, 0.f, 0.f};
    v8f acc[8];
    #pragma unroll
    for (int c = 0; c < 8; ++c) acc[c] = zero;

    const int base0 = laneHi * 8;           // K run for e = 0..7
    const int base1 = 16 + laneHi * 8;      // K run for e = 8..15

    for (int kc = 0; kc < NN / 32; ++kc) {
        const int k0 = kc * 32;

        union { int4 q[4]; int s[16]; } au;
        const int4* ap = (const int4*)(arow + k0);
        au.q[0] = ap[base0 >> 2];
        au.q[1] = ap[(base0 >> 2) + 1];
        au.q[2] = ap[base1 >> 2];
        au.q[3] = ap[(base1 >> 2) + 1];

        union { float4 q[4]; float s[16]; } fu;
        const float4* fp = (const float4*)(f2 + k0);
        fu.q[0] = fp[base0 >> 2];
        fu.q[1] = fp[(base0 >> 2) + 1];
        fu.q[2] = fp[base1 >> 2];
        fu.q[3] = fp[(base1 >> 2) + 1];

        v16h A;
        #pragma unroll
        for (int e = 0; e < 16; ++e) {
            const float lt = lrelu(f1v + fu.s[e]);
            const float p  = __expf(lt - mv) * invs;
            A[e] = (_Float16)(au.s[e] > 0 ? p : 0.f);
        }

        #pragma unroll
        for (int c = 0; c < 8; ++c) {
            const v16h B = bmat[(size_t)(kc * 8 + c) * 32 + lane];
            acc[c] = __builtin_amdgcn_wmma_f32_16x16x32_f16(
                false, A, false, B, (short)0, acc[c], false, false);
        }
    }

    // C/D layout: VGPR r holds M = r + 8*laneHi, N = lane%16 (ISA 7.12.2)
    #pragma unroll
    for (int c = 0; c < 8; ++c) {
        #pragma unroll
        for (int r = 0; r < 8; ++r) {
            const int row = i0 + r + laneHi * 8;
            const int col = c * 16 + laneLo;
            hprime[(size_t)row * FF + col] = acc[c][r];
        }
    }
}

// ---------------------------------------------------------------------------
// Kernel 4: out[f] = elu( sum_i v[i] * x[i,f] ), f = 0..127
// ---------------------------------------------------------------------------
__global__ void gat_out(const float* __restrict__ x, const float* __restrict__ vrow,
                        float* __restrict__ out) {
    __shared__ float red[256];
    const int f = blockIdx.x, t = threadIdx.x;
    float acc = 0.f;
    for (int i = t; i < NN; i += 256)
        acc += vrow[i] * x[(size_t)i * FF + f];
    red[t] = acc;
    __syncthreads();
    for (int off = 128; off > 0; off >>= 1) {
        if (t < off) red[t] += red[t + off];
        __syncthreads();
    }
    if (t == 0) {
        const float tv = red[0];
        out[f] = tv > 0.f ? tv : __expf(tv) - 1.f;
    }
}

extern "C" void kernel_launch(void* const* d_in, const int* in_sizes, int n_in,
                              void* d_out, int out_size, void* d_ws, size_t ws_size,
                              hipStream_t stream) {
    (void)in_sizes; (void)n_in; (void)out_size; (void)ws_size;
    const float* x   = (const float*)d_in[0];
    const int*   adj = (const int*)d_in[1];
    const float* W   = (const float*)d_in[2];
    const float* a   = (const float*)d_in[3];
    const float* W2  = (const float*)d_in[4];

    float* out    = (float*)d_out;      // [128] elu output
    float* hprime = out + FF;           // [8192 x 128] h_prime

    // workspace carve (~2.26 MB)
    char* ws = (char*)d_ws;
    _Float16* hB = (_Float16*)ws;                         // 8192*128*2 B
    float* f1   = (float*)(ws + (size_t)NN * FF * 2);     // 8192 floats each
    float* f2   = f1 + NN;
    float* mrow = f2 + NN;
    float* srow = mrow + NN;
    float* vrow = srow + NN;

    gat_prep     <<<NN,      FF,  0, stream>>>(x, W, a, hB, f1, f2);
    gat_stats    <<<NN,      256, 0, stream>>>(adj, W2, f1, f2, mrow, srow, vrow);
    gat_attn_gemm<<<NN / 16, 32,  0, stream>>>(adj, hB, f1, f2, mrow, srow, hprime);
    gat_out      <<<FF,      256, 0, stream>>>(x, vrow, out);
}
